// Var_Net_4492535791757
// MI455X (gfx1250) — matrix operand order repaired and checked by reading, not compiled, not run on hardware
//
#include <hip/hip_runtime.h>
#include <hip/hip_bf16.h>

// ---------------------------------------------------------------------------
// Var_Net: sigma = rsqrt( exp(-(||z-c||^2) * lam) @ (W*W) + EPS )
// B=262144, K=64, DZ=128, OUT=16.  gfx1250 wave32 WMMA bf16 path.
// ---------------------------------------------------------------------------

#define EPS_V 1e-06f
#define LOG2E 1.4426950408889634f

typedef __attribute__((ext_vector_type(16))) __bf16 v16bf;
typedef __attribute__((ext_vector_type(8)))  float  v8f;

union BF16x16 { v16bf v; unsigned int u[8]; };

// round-to-nearest-even f32 -> bf16 (as u16), bit version (prep kernel only)
__device__ __forceinline__ unsigned short f2bf16h(float f) {
    unsigned int x = __float_as_uint(f);
    return (unsigned short)((x + 0x7FFFu + ((x >> 16) & 1u)) >> 16);
}

// pack two f32 into one dword of bf16 pairs (lo = even k, hi = odd k)
#if __has_builtin(__builtin_amdgcn_cvt_pk_bf16_f32)
__device__ __forceinline__ unsigned int bfpack(float lo, float hi) {
    union {
        decltype(__builtin_amdgcn_cvt_pk_bf16_f32(0.f, 0.f)) v;
        unsigned int u;
    } c;
    c.v = __builtin_amdgcn_cvt_pk_bf16_f32(lo, hi);   // v_cvt_pk_bf16_f32
    return c.u;
}
__device__ __forceinline__ unsigned short f2bf_store(float f) {
    return (unsigned short)bfpack(f, f);
}
#else
// round-half-up + v_perm_b32 pack: 3 VALU per pair
__device__ __forceinline__ unsigned int bfpack(float lo, float hi) {
    unsigned int a = __float_as_uint(lo) + 0x8000u;
    unsigned int b = __float_as_uint(hi) + 0x8000u;
    // dst = { b[31:16], a[31:16] }  (bytes of {S0=b, S1=a}: 7,6 | 3,2)
    return __builtin_amdgcn_perm(b, a, 0x07060302u);
}
// round-half-up; store high half (compiler emits ds_store_b16_d16_hi)
__device__ __forceinline__ unsigned short f2bf_store(float f) {
    return (unsigned short)((__float_as_uint(f) + 0x8000u) >> 16);
}
#endif

__device__ __forceinline__ float lane_read(float v, int srcLane) {
    return __int_as_float(
        __builtin_amdgcn_ds_bpermute(srcLane << 2, __float_as_int(v)));
}

// Workspace layout (bytes):
//   [0,      16384) : cnts as bf16, row-major [64][128]
//   [16384,  16640) : c2[64]  (f32)  = ||center||^2
//   [16640,  18688) : WWT bf16, row-major [16][64]  (WWT[j][k] = W[k][j]^2)
//   [18688,  18944) : pl[64]  (f32)  = lam[n] * log2(e)

__global__ void varnet_prep(const float* __restrict__ cnts,
                            const float* __restrict__ lams,
                            const float* __restrict__ W,
                            unsigned char* __restrict__ ws) {
    const int tid = threadIdx.x;
    unsigned short* cb  = (unsigned short*)ws;
    float*          c2  = (float*)(ws + 16384);
    unsigned short* wwt = (unsigned short*)(ws + 16640);
    float*          pl  = (float*)(ws + 18688);

    for (int i = tid; i < 64 * 128; i += 256) cb[i] = f2bf16h(cnts[i]);

    if (tid < 64) {
        const float* c = cnts + tid * 128;
        float s = 0.f;
        for (int k = 0; k < 128; ++k) s += c[k] * c[k];
        c2[tid] = s;
        pl[tid] = lams[tid] * LOG2E;
    }
    for (int i = tid; i < 16 * 64; i += 256) {
        int j = i >> 6, k = i & 63;
        float w = W[k * 16 + j];
        wwt[i] = f2bf16h(w * w);
    }
}

// 16-byte int vector types with explicit address spaces for the async builtin
typedef int v4i_vs __attribute__((vector_size(16)));
typedef __attribute__((address_space(1))) v4i_vs gv4i;   // global
typedef __attribute__((address_space(3))) v4i_vs lv4i;   // LDS

__global__ __launch_bounds__(256) void varnet_main(
    const float* __restrict__ z,
    const unsigned char* __restrict__ ws,
    float* __restrict__ out) {

    // [0,16384): cnts bf16 staged; [16384 + wave*2048): per-wave V scratch (bf16 16x64)
    __shared__ __align__(16) unsigned char lds[16384 + 8 * 2048];

    const int tid  = threadIdx.x;
    const int lane = tid & 31;
    const int wave = tid >> 5;
    const int m    = lane & 15;   // row within 16-row tile / B-col / out-col
    const int h    = lane >> 4;   // lane half

    // ---- stage cnts_bf16 (16 KB) into LDS via async DMA path ----
#if __has_builtin(__builtin_amdgcn_global_load_async_to_lds_b128)
    {
        #pragma unroll
        for (int i = 0; i < 4; ++i) {
            gv4i* g = (gv4i*)(ws + (tid + 256 * i) * 16);
            lv4i* l = (lv4i*)(lds + (tid + 256 * i) * 16);
            __builtin_amdgcn_global_load_async_to_lds_b128(g, l, 0, 0);
        }
#if __has_builtin(__builtin_amdgcn_s_wait_asynccnt)
        __builtin_amdgcn_s_wait_asynccnt(0);
#else
        asm volatile("s_wait_asynccnt 0x0" ::: "memory");
#endif
    }
#else
    {
        const uint4* src = (const uint4*)ws;
        uint4*       dst = (uint4*)lds;
        #pragma unroll
        for (int i = 0; i < 4; ++i) dst[tid + 256 * i] = src[tid + 256 * i];
    }
#endif
    __syncthreads();

    const long rowBase = (long)blockIdx.x * 128 + (long)wave * 16;
    const long row     = rowBase + m;

    // ---- load 16x128 z tile in WMMA A-layout (bf16), accumulate ||z||^2 ----
    // A 16x32 bf16 layout: lane(m,h); VGPR j 0..3: k = 32s + 16h + 2j(+1)
    //                                 VGPR j 4..7: k = 32s + 16h + 8 + 2(j-4)(+1)
    unsigned int apk[4][8];
    float z2p = 0.f;
    const float* zrow = z + row * 128;
    #pragma unroll
    for (int s = 0; s < 4; ++s) {
        #pragma unroll
        for (int r = 0; r < 2; ++r) {
            const int kb = 32 * s + 8 * r + 16 * h;
            float4 a0 = *(const float4*)(zrow + kb);
            float4 a1 = *(const float4*)(zrow + kb + 4);
            z2p += a0.x * a0.x + a0.y * a0.y + a0.z * a0.z + a0.w * a0.w;
            z2p += a1.x * a1.x + a1.y * a1.y + a1.z * a1.z + a1.w * a1.w;
            apk[s][4 * r + 0] = bfpack(a0.x, a0.y);
            apk[s][4 * r + 1] = bfpack(a0.z, a0.w);
            apk[s][4 * r + 2] = bfpack(a1.x, a1.y);
            apk[s][4 * r + 3] = bfpack(a1.z, a1.w);
        }
    }
    // complete row norms (other half of k lives in lane^16) and redistribute
    // to C-layout rows (row index of vgpr v at this lane = v + 8h)
    float z2row = z2p + lane_read(z2p, lane ^ 16);
    float z2v[8];
    #pragma unroll
    for (int v = 0; v < 8; ++v) z2v[v] = lane_read(z2row, v + 8 * h);

    // ---- preload WWT B-tiles (B 32x16 bf16 layout: col=m, k runs of 16 by half) ----
    unsigned int wwb[2][8];
    {
        const unsigned char* wwt = ws + 16640;
        #pragma unroll
        for (int s = 0; s < 2; ++s) {
            const int kb = 32 * s + 16 * h;
            uint4 q0 = *(const uint4*)(wwt + m * 128 + 2 * kb);
            uint4 q1 = *(const uint4*)(wwt + m * 128 + 2 * kb + 16);
            wwb[s][0] = q0.x; wwb[s][1] = q0.y; wwb[s][2] = q0.z; wwb[s][3] = q0.w;
            wwb[s][4] = q1.x; wwb[s][5] = q1.y; wwb[s][6] = q1.z; wwb[s][7] = q1.w;
        }
    }

    unsigned char* vlds = lds + 16384 + wave * 2048;
    const float*   c2p  = (const float*)(ws + 16384);
    const float*   plp  = (const float*)(ws + 18688);

    // ---- GEMM1: G = z_tile @ cnts^T  (4 N-tiles x 4 K-steps) then V=exp ----
    #pragma unroll
    for (int t = 0; t < 4; ++t) {
        const int n = 16 * t + m;        // center index for this lane's B column
        v8f acc = {};
        #pragma unroll
        for (int s = 0; s < 4; ++s) {
            const int kb = 32 * s + 16 * h;
            const unsigned char* bp = lds + n * 256 + 2 * kb;
            uint4 q0 = *(const uint4*)bp;
            uint4 q1 = *(const uint4*)(bp + 16);
            BF16x16 bv;
            bv.u[0] = q0.x; bv.u[1] = q0.y; bv.u[2] = q0.z; bv.u[3] = q0.w;
            bv.u[4] = q1.x; bv.u[5] = q1.y; bv.u[6] = q1.z; bv.u[7] = q1.w;
            BF16x16 av;
            #pragma unroll
            for (int j = 0; j < 8; ++j) av.u[j] = apk[s][j];
            acc = __builtin_amdgcn_wmma_f32_16x16x32_bf16(
                false, av.v, false, bv.v, (short)0, acc, false, false);
        }
        const float c2n = c2p[n];
        const float pln = plp[n];
        #pragma unroll
        for (int v = 0; v < 8; ++v) {
            // exponent = (2*G - (||z||^2 + ||c||^2)) * lam * log2(e)
            float zc = z2v[v] + c2n;
            float e  = __builtin_fmaf(acc[v], 2.0f, -zc) * pln;
            float Vx = __builtin_amdgcn_exp2f(e);              // v_exp_f32
            // per-wave V scratch, row-major bf16 [16][64]; C layout: row=v+8h, col=n
            *(unsigned short*)(vlds + (v + 8 * h) * 128 + 2 * n) = f2bf_store(Vx);
        }
    }

    // ---- GEMM2: beta = V @ (W*W) + EPS  (K=64 -> 2 K-steps), sigma = rsqrt ----
    v8f bacc;
    #pragma unroll
    for (int v = 0; v < 8; ++v) bacc[v] = EPS_V;
    #pragma unroll
    for (int s = 0; s < 2; ++s) {
        BF16x16 av;
        #pragma unroll
        for (int r = 0; r < 2; ++r) {
            const int kb = 32 * s + 8 * r + 16 * h;
            uint4 q = *(const uint4*)(vlds + m * 128 + 2 * kb);
            av.u[4 * r + 0] = q.x; av.u[4 * r + 1] = q.y;
            av.u[4 * r + 2] = q.z; av.u[4 * r + 3] = q.w;
        }
        BF16x16 bv;
        #pragma unroll
        for (int j = 0; j < 8; ++j) bv.u[j] = wwb[s][j];
        bacc = __builtin_amdgcn_wmma_f32_16x16x32_bf16(
            false, av.v, false, bv.v, (short)0, bacc, false, false);
    }

    #pragma unroll
    for (int v = 0; v < 8; ++v) {
        float sig = __builtin_amdgcn_rsqf(bacc[v]);          // v_rsq_f32
        out[(rowBase + v + 8 * h) * 16 + m] = sig;
    }
}

extern "C" void kernel_launch(void* const* d_in, const int* in_sizes, int n_in,
                              void* d_out, int out_size, void* d_ws, size_t ws_size,
                              hipStream_t stream) {
    const float* z    = (const float*)d_in[0];   // [B,128]
    const float* cnts = (const float*)d_in[1];   // [64,128]
    const float* lams = (const float*)d_in[2];   // [64]
    const float* W    = (const float*)d_in[3];   // [64,16]
    float*       out  = (float*)d_out;           // [B,16]
    unsigned char* ws = (unsigned char*)d_ws;

    const int B      = in_sizes[0] / 128;        // 262144
    const int blocks = B / 128;                  // 128 rows per block (8 waves x 16)

    varnet_prep<<<1, 256, 0, stream>>>(cnts, lams, W, ws);
    varnet_main<<<blocks, 256, 0, stream>>>(z, ws, out);
}
